// OctonionHeadMixer_6640019440341
// MI455X (gfx1250) — matrix operand order repaired and checked by reading, not compiled, not run on hardware
//
#include <hip/hip_runtime.h>

// ---------------------------------------------------------------------------
// OctonionHeadMixer for MI455X (gfx1250, wave32, WMMA)
//
// out[b,i,t,e] = sum_{j,d} x[b,j,t,d] * SGN[i][j] * W[i^j][d,e] * beta[e]
//  -> one big GEMM: (B*T=65536 x 512) @ Wbig(512 x 512)
// Split-precision bf16 (hi + lo) with f32 accumulate via
// v_wmma_f32_16x16x32_bf16; Wbig prepacked into the WMMA B-operand layout.
// ---------------------------------------------------------------------------

typedef __attribute__((ext_vector_type(16))) __bf16 v16bf;
typedef __attribute__((ext_vector_type(8)))  __bf16 v8bf;
typedef __attribute__((ext_vector_type(8)))  float  v8f;
typedef __attribute__((ext_vector_type(4)))  float  v4f;

#define K_TOT    512
#define N_TOT    512
#define KSTEPS   16          // K_TOT / 32
#define M_TILE   64
#define LDS_K    520         // 512 + 8 bf16 pad per row (bank spread)
#define LDS_BYTES (2 * M_TILE * LDS_K * 2)   // Ahi + Alo, bf16

// Octonion sign table: SGN[i][j] such that e_j * e_{i^j} = SGN[i][j] * e_i
// (derived from the reference's Cayley-Dickson construction; widx[i][j]==i^j)
__device__ const signed char SGN_TAB[8][8] = {
    { 1,-1,-1,-1,-1,-1,-1,-1},
    { 1, 1, 1,-1, 1,-1,-1, 1},
    { 1,-1, 1, 1, 1, 1,-1,-1},
    { 1, 1,-1, 1, 1,-1, 1,-1},
    { 1,-1,-1,-1, 1, 1, 1, 1},
    { 1, 1,-1, 1,-1, 1,-1, 1},
    { 1, 1, 1,-1,-1,-1, 1,-1},
    { 1,-1, 1, 1,-1,-1,-1, 1}
};

static __device__ __forceinline__ __bf16 us_to_bf16(unsigned short v) {
    union { unsigned short u; __bf16 b; } c; c.u = v; return c.b;
}

// fp32 -> bf16 hi (RNE) + bf16 lo (residual, RNE)
static __device__ __forceinline__ void split_bf16(float x, unsigned short& h,
                                                  unsigned short& l) {
    unsigned u  = __float_as_uint(x);
    unsigned rh = u + 0x7FFFu + ((u >> 16) & 1u);
    h = (unsigned short)(rh >> 16);
    float hf = __uint_as_float(((unsigned)h) << 16);
    float lo = x - hf;
    unsigned ul = __float_as_uint(lo);
    unsigned rl = ul + 0x7FFFu + ((ul >> 16) & 1u);
    l = (unsigned short)(rl >> 16);
}

// ---------------------------------------------------------------------------
// Pack Wbig (with signs + beta folded) into WMMA B-operand order:
//   bp[(kk*512 + n)*32 + kv]  for k = kk*32 + kv, column n = i*64 + e
// Lane n%16 of the B operand then reads 16 contiguous bf16 at khalf*16.
// ---------------------------------------------------------------------------
__global__ __launch_bounds__(256) void octo_pack_w(
    const float* __restrict__ W, const float* __restrict__ beta,
    __bf16* __restrict__ bph, __bf16* __restrict__ bpl) {
    int idx = blockIdx.x * 256 + threadIdx.x;   // 0 .. 512*512-1
    int n = idx & 511;
    int k = idx >> 9;
    int i = n >> 6, e = n & 63;
    int j = k >> 6, d = k & 63;
    int a = i ^ j;
    float val = (float)SGN_TAB[i][j] * W[(a << 12) + (d << 6) + e] * beta[e];
    unsigned short h, l;
    split_bf16(val, h, l);
    int off = (((k >> 5) * 512 + n) << 5) + (k & 31);
    bph[off] = us_to_bf16(h);
    bpl[off] = us_to_bf16(l);
}

// ---------------------------------------------------------------------------
// Main GEMM: one block = 64 tokens x full N=512, K=512.
// 8 waves: wave = wm(0..3) x wn(0..1); wave tile = 16 M x 256 N.
// ---------------------------------------------------------------------------
__global__ __launch_bounds__(256) void octo_gemm(
    const float* __restrict__ x, const __bf16* __restrict__ bph,
    const __bf16* __restrict__ bpl, float* __restrict__ out) {
    extern __shared__ __align__(32) char smem_raw[];
    __bf16* Ahi = (__bf16*)smem_raw;
    __bf16* Alo = Ahi + M_TILE * LDS_K;

    const int tid  = threadIdx.x;
    const int mblk = blockIdx.x;              // 0..1023
    const int b    = mblk >> 6;               // 4096/64 = 64 tiles per batch
    const int t0   = (mblk & 63) << 6;
    const float* xbase = x + (size_t)b * 2097152 + (size_t)t0 * 64;

    // ---- Phase 1: stage 64x512 fp32 x-tile as bf16 hi/lo in LDS ----
    #pragma unroll 4
    for (int it = 0; it < 32; ++it) {
        int g   = tid + (it << 8);            // 0..8191 float4 groups
        int row = g >> 7;                     // token row 0..63
        int kq  = g & 127;
        int k   = kq << 2;                    // k = j*64 + d, d%4==0
        int j   = k >> 6, d = k & 63;
        v4f v = *(const v4f*)(xbase + j * 262144 + row * 64 + d);
        int lb = row * LDS_K + k;
        #pragma unroll
        for (int c = 0; c < 4; ++c) {
            unsigned short h, l;
            split_bf16(v[c], h, l);
            Ahi[lb + c] = us_to_bf16(h);
            Alo[lb + c] = us_to_bf16(l);
        }
    }
    __syncthreads();

    // ---- Phase 2: WMMA mainloop ----
    const int wave  = tid >> 5;
    const int lane  = tid & 31;
    const int wm    = wave & 3;               // M sub-tile 0..3
    const int wn    = wave >> 2;              // N half 0..1
    const int khalf = lane >> 4;              // K-half select per A/B layout
    const int mrow  = (wm << 4) + (lane & 15);
    const int coln  = (wn << 8) + (lane & 15);

    v8f acc[16] = {};                         // 16 tiles of 16x16 f32

    for (int kk = 0; kk < KSTEPS; ++kk) {
        const int kb = kk << 5;
        // A fragment: lanes<16 hold K 0..15, lanes>=16 hold K 16..31
        const __bf16* ah = Ahi + mrow * LDS_K + kb + (khalf << 4);
        const __bf16* al = Alo + mrow * LDS_K + kb + (khalf << 4);
        v8bf ah0 = *(const v8bf*)(ah);
        v8bf ah1 = *(const v8bf*)(ah + 8);
        v8bf al0 = *(const v8bf*)(al);
        v8bf al1 = *(const v8bf*)(al + 8);
        v16bf ahi_f = __builtin_shufflevector(ah0, ah1, 0,1,2,3,4,5,6,7,
                                              8,9,10,11,12,13,14,15);
        v16bf alo_f = __builtin_shufflevector(al0, al1, 0,1,2,3,4,5,6,7,
                                              8,9,10,11,12,13,14,15);
        #pragma unroll
        for (int nt = 0; nt < 16; ++nt) {
            const int col = coln + (nt << 4);
            const size_t boff = ((size_t)(kk * 512 + col) << 5) + (khalf << 4);
            v16bf bh = *(const v16bf*)(bph + boff);
            v16bf bl = *(const v16bf*)(bpl + boff);
            acc[nt] = __builtin_amdgcn_wmma_f32_16x16x32_bf16(
                false, ahi_f, false, bh, (short)0, acc[nt], false, false);
            acc[nt] = __builtin_amdgcn_wmma_f32_16x16x32_bf16(
                false, ahi_f, false, bl, (short)0, acc[nt], false, false);
            acc[nt] = __builtin_amdgcn_wmma_f32_16x16x32_bf16(
                false, alo_f, false, bh, (short)0, acc[nt], false, false);
        }
    }

    // ---- Store: C/D layout -> lanes<16 rows 0..7, lanes>=16 rows 8..15 ----
    const int trow0 = t0 + (wm << 4) + (khalf << 3);
    #pragma unroll
    for (int nt = 0; nt < 16; ++nt) {
        const int n = (wn << 8) + (nt << 4) + (lane & 15);
        const int i = n >> 6, e = n & 63;
        float* o = out + (size_t)b * 2097152 + (size_t)i * 262144
                       + (size_t)trow0 * 64 + e;
        #pragma unroll
        for (int r = 0; r < 8; ++r) o[r * 64] = acc[nt][r];
    }
}

// ---------------------------------------------------------------------------
extern "C" void kernel_launch(void* const* d_in, const int* in_sizes, int n_in,
                              void* d_out, int out_size, void* d_ws,
                              size_t ws_size, hipStream_t stream) {
    (void)in_sizes; (void)n_in; (void)out_size; (void)ws_size;
    const float* x    = (const float*)d_in[0];
    const float* W    = (const float*)d_in[1];
    const float* beta = (const float*)d_in[2];
    float* out = (float*)d_out;

    __bf16* bph = (__bf16*)d_ws;              // 512*512 bf16 = 512 KB
    __bf16* bpl = bph + 512 * 512;            // another 512 KB

    octo_pack_w<<<1024, 256, 0, stream>>>(W, beta, bph, bpl);

    (void)hipFuncSetAttribute((const void*)octo_gemm,
                              hipFuncAttributeMaxDynamicSharedMemorySize,
                              LDS_BYTES);
    octo_gemm<<<1024, 256, LDS_BYTES, stream>>>(x, bph, bpl, out);
}
